// CausalSelfAttention_52407190945977
// MI455X (gfx1250) — matrix operand order repaired and checked by reading, not compiled, not run on hardware
//
#include <hip/hip_runtime.h>
#include <hip/hip_bf16.h>
#include <math.h>

// ---------------------------------------------------------------------------
// CausalSelfAttention for MI455X (gfx1250, wave32, WMMA).
//   B=4, T=2048, D=1024.  All matmuls via v_wmma_f32_16x16x32_f16.
// Pipeline:
//   K1 qkv : Qf16[8192][1024], Kf16[8192][1024], Vt f16[1024][8192] (transposed)
//   K2 attn: S f32[B][T][T] (raw, causal-masked) -> softmax -> P f16[B][T][T]
//   K3 pv  : out = P @ V  (K-loop bounded by causal frontier)
// Workspace layout (bytes): Qf16@0(16M) Kf16@16M Vt@32M P@48M(32M) S@80M(64M)
// S and P both fit in the 192MB L2 -> materialization beats recompute.
// ---------------------------------------------------------------------------

typedef _Float16 h16;
typedef __attribute__((ext_vector_type(16))) _Float16 v16h;
typedef __attribute__((ext_vector_type(8)))  _Float16 v8h;
typedef __attribute__((ext_vector_type(8)))  float    v8f;

#define TT    2048
#define DD    1024
#define MTOT  8192   // B*T

__device__ __forceinline__ v8f vzero8() {
    v8f z = {0.f, 0.f, 0.f, 0.f, 0.f, 0.f, 0.f, 0.f};
    return z;
}

// A fragment (16x32 f16, M x K). Lane L: row = base_row + (L&15), h = L>>4.
// v16h elems 0..7  -> K = 8h + 0..7 ;  elems 8..15 -> K = 16 + 8h + 0..7.
__device__ __forceinline__ v16h load_a_frag(const h16* tile, int row0, int lda, int lane) {
    int m = row0 + (lane & 15);
    int h = lane >> 4;
    const h16* p = tile + m * lda + h * 8;
    v8h lo = *(const v8h*)p;
    v8h hi = *(const v8h*)(p + 16);
    return __builtin_shufflevector(lo, hi, 0,1,2,3,4,5,6,7,8,9,10,11,12,13,14,15);
}

// B fragment (32x16 f16, K x N) from an LDS tile stored as [N][K].
// Lane L: col = col0 + (L&15), h = L>>4; elems 0..7 -> K = 8h + 0..7 ; elems 8..15 -> K = 16 + 8h + 0..7.
__device__ __forceinline__ v16h load_b_frag(const h16* tile, int col0, int ldb, int lane) {
    int n = col0 + (lane & 15);
    const h16* p = tile + n * ldb + (lane >> 4) * 8;
    v8h lo = *(const v8h*)p;
    v8h hi = *(const v8h*)(p + 16);
    return __builtin_shufflevector(lo, hi, 0,1,2,3,4,5,6,7,8,9,10,11,12,13,14,15);
}

__device__ __forceinline__ v8f wmma16(v16h a, v16h b, v8f c) {
    return __builtin_amdgcn_wmma_f32_16x16x32_f16(false, a, false, b, (short)0, c, false, false);
}

// ---------------------------------------------------------------------------
// Kernel 1: fused QKV projection. out[m][n] = sum_d x[m][d] * W[n][d]
// N in [0,3072): 0..1023 -> Wq, 1024..2047 -> Wk, 2048..3071 -> Wv.
// Block 128 thr = 4 waves; block tile 64x64; wave tile 32x32 (2x2 WMMA).
// V epilogue transposes 64x64 tile through LDS -> coalesced v8h stores.
// ---------------------------------------------------------------------------
__global__ void __launch_bounds__(128)
qkv_kernel(const float* __restrict__ x,
           const float* __restrict__ Wq, const float* __restrict__ Wk,
           const float* __restrict__ Wv,
           h16* __restrict__ Qf, h16* __restrict__ Kf, h16* __restrict__ Vt) {
    __shared__ h16 As[64 * 40];   // x tile     [64 rows][32 k] pad->40
    __shared__ h16 Bs[64 * 40];   // weight tile[64 n  ][32 k] pad->40
    __shared__ h16 Cs[64 * 72];   // V transpose staging [n_local][m_local] pad->72

    const int tid  = threadIdx.x;
    const int lane = tid & 31;
    const int w    = tid >> 5;
    const int wm   = w >> 1;      // 0..1
    const int wn   = w & 1;       // 0..1
    const int m0   = blockIdx.y * 64;
    const int n0   = blockIdx.x * 64;
    const int mat  = n0 >> 10;            // 0=Q 1=K 2=V
    const int nn0  = n0 & 1023;           // col within the selected matrix
    const float* __restrict__ W = (mat == 0) ? Wq : ((mat == 1) ? Wk : Wv);

    v8f acc[2][2];
    acc[0][0] = vzero8(); acc[0][1] = vzero8();
    acc[1][0] = vzero8(); acc[1][1] = vzero8();

    for (int k0 = 0; k0 < DD; k0 += 32) {
        // stage A: 64x32 floats of x -> f16 LDS.  512 float4 / 128 thr = 4 each
        #pragma unroll
        for (int i = 0; i < 4; ++i) {
            int lin = tid + i * 128;
            int r = lin >> 3, c4 = lin & 7;
            float4 v = *(const float4*)(x + (size_t)(m0 + r) * DD + k0 + c4 * 4);
            h16* dst = &As[r * 40 + c4 * 4];
            dst[0] = (h16)v.x; dst[1] = (h16)v.y; dst[2] = (h16)v.z; dst[3] = (h16)v.w;
        }
        // stage B: 64x32 floats of W -> f16 LDS
        #pragma unroll
        for (int i = 0; i < 4; ++i) {
            int lin = tid + i * 128;
            int r = lin >> 3, c4 = lin & 7;
            float4 v = *(const float4*)(W + (size_t)(nn0 + r) * DD + k0 + c4 * 4);
            h16* dst = &Bs[r * 40 + c4 * 4];
            dst[0] = (h16)v.x; dst[1] = (h16)v.y; dst[2] = (h16)v.z; dst[3] = (h16)v.w;
        }
        __syncthreads();

        v16h a0 = load_a_frag(As, wm * 32 +  0, 40, lane);
        v16h a1 = load_a_frag(As, wm * 32 + 16, 40, lane);
        v16h b0 = load_b_frag(Bs, wn * 32 +  0, 40, lane);
        v16h b1 = load_b_frag(Bs, wn * 32 + 16, 40, lane);
        acc[0][0] = wmma16(a0, b0, acc[0][0]);
        acc[0][1] = wmma16(a0, b1, acc[0][1]);
        acc[1][0] = wmma16(a1, b0, acc[1][0]);
        acc[1][1] = wmma16(a1, b1, acc[1][1]);
        __syncthreads();
    }

    // Epilogue: D layout -> lane L holds col (L&15), rows r + 8*(L>>4).
    const int nl = lane & 15, hf = lane >> 4;
    if (mat == 2) {
        // transpose through LDS: Cs[n_local][m_local]; each lane's 8 accum
        // values are contiguous in m -> vectorizable ds_store.
        #pragma unroll
        for (int tm = 0; tm < 2; ++tm) {
            #pragma unroll
            for (int tn = 0; tn < 2; ++tn) {
                int nloc  = wn * 32 + tn * 16 + nl;
                int mbase = wm * 32 + tm * 16 + hf * 8;
                h16* dst = &Cs[nloc * 72 + mbase];
                #pragma unroll
                for (int r = 0; r < 8; ++r) dst[r] = (h16)acc[tm][tn][r];
            }
        }
        __syncthreads();
        // cooperative coalesced store: 64 rows x 64 halves = 512 v8h chunks
        #pragma unroll
        for (int i = 0; i < 4; ++i) {
            int lin = tid + i * 128;
            int r = lin >> 3, c = lin & 7;
            *(v8h*)&Vt[(size_t)(nn0 + r) * MTOT + m0 + c * 8] =
                *(const v8h*)&Cs[r * 72 + c * 8];
        }
    } else {
        h16* __restrict__ O = (mat == 0) ? Qf : Kf;
        #pragma unroll
        for (int tm = 0; tm < 2; ++tm) {
            #pragma unroll
            for (int tn = 0; tn < 2; ++tn) {
                int n = nn0 + wn * 32 + tn * 16 + nl;
                #pragma unroll
                for (int r = 0; r < 8; ++r) {
                    int m = m0 + wm * 32 + tm * 16 + hf * 8 + r;
                    O[(size_t)m * DD + n] = (h16)acc[tm][tn][r];
                }
            }
        }
    }
}

// ---------------------------------------------------------------------------
// Kernel 2: scores + softmax for one (batch, 32-query tile).
// 128 thr = 4 waves = 2 row-frags x 2 key-strips.  Q tile + 32-key tile in LDS
// (~130KB total: safe under the 160KB CU-mode ceiling).
// Phase 1: S = Q K^T via WMMA, causal mask, store raw f32, track row max.
// Phase 2: scalar sweeps over stored S -> P = softmax (f16).
// ---------------------------------------------------------------------------
#define LDQ 1032   // 1024 + 8 pad (halves)
#define QS_BYTES    (32 * LDQ * 2)        // 66048
#define KS_BYTES    (32 * LDQ * 2)        // 66048
#define SMEM2_BYTES (QS_BYTES + KS_BYTES + (64 + 32 + 128) * 4)

__global__ void __launch_bounds__(128)
attn_kernel(const h16* __restrict__ Qf, const h16* __restrict__ Kf,
            float* __restrict__ Sbuf, h16* __restrict__ Pbuf) {
    extern __shared__ char smem[];
    h16*   Qs     = (h16*)smem;                           // [32][LDQ]
    h16*   Ks     = (h16*)(smem + QS_BYTES);              // [32][LDQ]
    float* redbuf = (float*)(smem + QS_BYTES + KS_BYTES); // [32][2] partial max
    float* rowmax = redbuf + 64;                          // [32]
    float* rowsum = rowmax + 32;                          // [32][4]

    const int tid  = threadIdx.x;
    const int lane = tid & 31;
    const int w    = tid >> 5;
    const int rf   = w >> 1;   // 0..1 : q rows q0 + rf*16
    const int ks   = w & 1;    // 0..1 : keys j0 + ks*16
    const int q0   = blockIdx.x * 32;
    const int b    = blockIdx.y;
    const int nl   = lane & 15, hf = lane >> 4;
    const int jend = q0 + 32;                  // causal frontier for this tile

    // stage Q tile: 32 x 1024 halves, 4096 v8h chunks / 128 thr = 32 each
    #pragma unroll 4
    for (int i = 0; i < 32; ++i) {
        int lin = tid + i * 128;
        int r = lin >> 7, c = lin & 127;
        *(v8h*)&Qs[r * LDQ + c * 8] =
            *(const v8h*)&Qf[(size_t)(b * TT + q0 + r) * DD + c * 8];
    }

    float m_run[8];
    #pragma unroll
    for (int r = 0; r < 8; ++r) m_run[r] = -3.0e38f;

    for (int j0 = 0; j0 < jend; j0 += 32) {
        __syncthreads();
        // stage K tile: 32 x 1024 halves / 128 thr = 32 chunks each
        #pragma unroll 4
        for (int i = 0; i < 32; ++i) {
            int lin = tid + i * 128;
            int r = lin >> 7, c = lin & 127;
            *(v8h*)&Ks[r * LDQ + c * 8] =
                *(const v8h*)&Kf[(size_t)(b * TT + j0 + r) * DD + c * 8];
        }
        __syncthreads();

        v8f s = vzero8();
        #pragma unroll 4
        for (int d0 = 0; d0 < DD; d0 += 32) {
            v16h a  = load_a_frag(Qs + d0, rf * 16, LDQ, lane);
            v16h bb = load_b_frag(Ks + d0, ks * 16, LDQ, lane);
            s = wmma16(a, bb, s);
        }

        const int j = j0 + ks * 16 + nl;
        #pragma unroll
        for (int r = 0; r < 8; ++r) {
            int q = q0 + rf * 16 + hf * 8 + r;
            float sv = (j <= q) ? s[r] : -1.0e30f;   // causal mask
            Sbuf[(size_t)(b * TT + q) * TT + j] = sv;
            m_run[r] = fmaxf(m_run[r], sv);
        }
    }

    // reduce row max across the 16 lanes holding the same row
    #pragma unroll
    for (int r = 0; r < 8; ++r) {
        float v = m_run[r];
        v = fmaxf(v, __shfl_xor(v, 1));
        v = fmaxf(v, __shfl_xor(v, 2));
        v = fmaxf(v, __shfl_xor(v, 4));
        v = fmaxf(v, __shfl_xor(v, 8));
        m_run[r] = v;
    }
    if (nl == 0) {
        #pragma unroll
        for (int r = 0; r < 8; ++r)
            redbuf[(rf * 16 + hf * 8 + r) * 2 + ks] = m_run[r];
    }
    __syncthreads();
    if (tid < 32) rowmax[tid] = fmaxf(redbuf[tid * 2], redbuf[tid * 2 + 1]);
    __syncthreads();

    // Phase 2: softmax.  4 threads per row (32 rows, 128 threads).
    const int row = tid >> 2, sub = tid & 3;
    const float mx = rowmax[row];
    const size_t sbase = (size_t)(b * TT + q0 + row) * TT;
    float lsum = 0.f;
    for (int j = sub; j < jend; j += 4) lsum += __expf(Sbuf[sbase + j] - mx);
    rowsum[row * 4 + sub] = lsum;
    __syncthreads();
    const float l = rowsum[row * 4 + 0] + rowsum[row * 4 + 1] +
                    rowsum[row * 4 + 2] + rowsum[row * 4 + 3];
    const float inv = 1.0f / l;
    for (int j = sub; j < jend; j += 4)
        Pbuf[sbase + j] = (h16)(__expf(Sbuf[sbase + j] - mx) * inv);
}

// ---------------------------------------------------------------------------
// Kernel 3: out[b] = P[b] (2048x2048) @ V[b] (2048x1024), K bounded causally.
// Block 128 thr = 4 waves; tile 64x64 like kernel 1.  B operand from Vt
// (d-major) so B fragments have contiguous K (= key index).
// ---------------------------------------------------------------------------
__global__ void __launch_bounds__(128)
pv_kernel(const h16* __restrict__ Pbuf, const h16* __restrict__ Vt,
          float* __restrict__ out) {
    __shared__ h16 As[64 * 40];   // P tile  [64 q][32 j]
    __shared__ h16 Bs[64 * 40];   // Vt tile [64 d][32 j]

    const int tid  = threadIdx.x;
    const int lane = tid & 31;
    const int w    = tid >> 5;
    const int wm   = w >> 1, wn = w & 1;
    const int n0   = blockIdx.x * 64;   // out cols (d)
    const int m0   = blockIdx.y * 64;   // q rows
    const int b    = blockIdx.z;
    const int kend = m0 + 64;           // causal frontier: keys < m0+64

    v8f acc[2][2];
    acc[0][0] = vzero8(); acc[0][1] = vzero8();
    acc[1][0] = vzero8(); acc[1][1] = vzero8();

    for (int j0 = 0; j0 < kend; j0 += 32) {
        // A: 64x32 halves of P  (256 v8h chunks / 128 thr = 2 each)
        #pragma unroll
        for (int i = 0; i < 2; ++i) {
            int lin = tid + i * 128;
            int r = lin >> 2, c = lin & 3;
            *(v8h*)&As[r * 40 + c * 8] =
                *(const v8h*)&Pbuf[(size_t)(b * TT + m0 + r) * TT + j0 + c * 8];
        }
        // B: 64x32 halves of Vt (rows = d, contiguous in token)
        #pragma unroll
        for (int i = 0; i < 2; ++i) {
            int lin = tid + i * 128;
            int r = lin >> 2, c = lin & 3;
            *(v8h*)&Bs[r * 40 + c * 8] =
                *(const v8h*)&Vt[(size_t)(n0 + r) * MTOT + b * TT + j0 + c * 8];
        }
        __syncthreads();

        v16h a0 = load_a_frag(As, wm * 32 +  0, 40, lane);
        v16h a1 = load_a_frag(As, wm * 32 + 16, 40, lane);
        v16h b0 = load_b_frag(Bs, wn * 32 +  0, 40, lane);
        v16h b1 = load_b_frag(Bs, wn * 32 + 16, 40, lane);
        acc[0][0] = wmma16(a0, b0, acc[0][0]);
        acc[0][1] = wmma16(a0, b1, acc[0][1]);
        acc[1][0] = wmma16(a1, b0, acc[1][0]);
        acc[1][1] = wmma16(a1, b1, acc[1][1]);
        __syncthreads();
    }

    const int nl = lane & 15, hf = lane >> 4;
    #pragma unroll
    for (int tm = 0; tm < 2; ++tm) {
        #pragma unroll
        for (int tn = 0; tn < 2; ++tn) {
            int n = n0 + wn * 32 + tn * 16 + nl;
            #pragma unroll
            for (int r = 0; r < 8; ++r) {
                int m = m0 + wm * 32 + tm * 16 + hf * 8 + r;
                out[(size_t)(b * TT + m) * DD + n] = acc[tm][tn][r];
            }
        }
    }
}

// ---------------------------------------------------------------------------
// Launch.  Workspace needed: 144 MiB (Q/K/Vt f16 + P f16 + S f32).
// ---------------------------------------------------------------------------
extern "C" void kernel_launch(void* const* d_in, const int* in_sizes, int n_in,
                              void* d_out, int out_size, void* d_ws, size_t ws_size,
                              hipStream_t stream) {
    const float* x  = (const float*)d_in[0];
    const float* Wq = (const float*)d_in[1];
    const float* Wk = (const float*)d_in[2];
    const float* Wv = (const float*)d_in[3];
    float* out = (float*)d_out;

    char* ws = (char*)d_ws;
    h16*   Qf   = (h16*)(ws + 0);                          // 16 MiB
    h16*   Kf   = (h16*)(ws + (size_t)16 * 1024 * 1024);   // 16 MiB
    h16*   Vt   = (h16*)(ws + (size_t)32 * 1024 * 1024);   // 16 MiB, [1024][8192]
    h16*   Pbuf = (h16*)(ws + (size_t)48 * 1024 * 1024);   // 32 MiB
    float* Sbuf = (float*)(ws + (size_t)80 * 1024 * 1024); // 64 MiB

    qkv_kernel<<<dim3(48, 128), 128, 0, stream>>>(x, Wq, Wk, Wv, Qf, Kf, Vt);
    attn_kernel<<<dim3(TT / 32, 4), 128, SMEM2_BYTES, stream>>>(Qf, Kf, Sbuf, Pbuf);
    pv_kernel<<<dim3(DD / 64, TT / 64, 4), 128, 0, stream>>>(Pbuf, Vt, out);
}
